// BiMambaAggregator_1460288881206
// MI455X (gfx1250) — matrix operand order repaired and checked by reading, not compiled, not run on hardware
//
#include <hip/hip_runtime.h>
#include <hip/hip_bf16.h>

// ---------------- Model constants (from the reference) ----------------
#define BB   4
#define SEQ  1024
#define DM_  512
#define DI_  1024
#define DS_  16
#define DC_  4
#define DTR_ 32
#define ROWS (BB*SEQ)          // 4096 token rows

typedef __attribute__((ext_vector_type(16))) __bf16 v16bf;
typedef __attribute__((ext_vector_type(8)))  __bf16 v8bf;
typedef __attribute__((ext_vector_type(4)))  __bf16 v4bf;
typedef __attribute__((ext_vector_type(8)))  float  v8f;

union V16 { v16bf v; v8bf h[2]; };

__device__ __forceinline__ v4bf cvt4(float4 f) {
  v4bf r;
  r[0] = (__bf16)f.x; r[1] = (__bf16)f.y;
  r[2] = (__bf16)f.z; r[3] = (__bf16)f.w;
  return r;
}

// ---------------- Activation epilogues ----------------
__device__ __forceinline__ float act_apply(float v, int act) {
  if (act == 1) {                       // exact GELU: 0.5 x (1 + erf(x/sqrt2))
    v = 0.5f * v * (1.0f + erff(v * 0.70710678118654752f));
  } else if (act == 2) {                // softplus
    v = (v > 20.0f) ? v : log1pf(__expf(v));
  } else if (act == 3) {                // tanh
    v = tanhf(v);
  }
  return v;
}

// ---------------- WMMA GEMM: C[M,N] = act(A[M,K(lda)] @ W[K,N] + bias) + res
// Tile 64x64, K-step 32, 256 threads = 8 wave32; each wave: 2x 16x16 accumulators.
// REQUIRES: M % 64 == 0, N % 64 == 0, K % 32 == 0 (true for every call here).
__global__ __launch_bounds__(256) void wmma_gemm_k(
    const float* __restrict__ A, int lda,
    const float* __restrict__ W,
    const float* __restrict__ bias,
    const float* __restrict__ res,
    float* __restrict__ C,
    int M, int N, int K, int act)
{
  // Rows padded to 40 elements (80 bytes: multiple of 16B keeps v8bf runs aligned).
  __shared__ __bf16 As [64][40];   // A tile, row-major  [m][k]
  __shared__ __bf16 BsT[64][40];   // B tile, TRANSPOSED [n][k]

  const int m0   = blockIdx.y * 64;
  const int n0   = blockIdx.x * 64;
  const int tid  = threadIdx.x;
  const int lane = tid & 31;
  const int wave = tid >> 5;
  const int wm   = wave & 3;      // wave row tile 0..3
  const int wn   = wave >> 2;     // wave col group 0..1 (covers 32 cols)
  const int lr   = lane & 15;
  const int hi   = lane >> 4;     // half-wave select

  // staging indices: each thread moves 8 contiguous elements of each tile
  const int ar = tid >> 2;          // A row   0..63
  const int ac = (tid & 3) * 8;     // A col   0,8,16,24
  const int br = tid >> 3;          // B row k 0..31
  const int bc = (tid & 7) * 8;     // B col n 0..56

  v8f acc0 = {};
  v8f acc1 = {};

  for (int k0 = 0; k0 < K; k0 += 32) {
    // ---- stage A tile (64x32) f32 -> bf16, vector loads + vector LDS stores
    {
      const float* ap = &A[(size_t)(m0 + ar) * lda + (k0 + ac)];
      float4 f0 = *(const float4*)ap;
      float4 f1 = *(const float4*)(ap + 4);
      *(v4bf*)&As[ar][ac]     = cvt4(f0);
      *(v4bf*)&As[ar][ac + 4] = cvt4(f1);
    }
    // ---- stage B tile (32x64) f32 -> bf16, stored transposed: BsT[n][k]
    {
      const float* bp = &W[(size_t)(k0 + br) * N + (n0 + bc)];
      float4 f0 = *(const float4*)bp;
      float4 f1 = *(const float4*)(bp + 4);
      BsT[bc + 0][br] = (__bf16)f0.x;
      BsT[bc + 1][br] = (__bf16)f0.y;
      BsT[bc + 2][br] = (__bf16)f0.z;
      BsT[bc + 3][br] = (__bf16)f0.w;
      BsT[bc + 4][br] = (__bf16)f1.x;
      BsT[bc + 5][br] = (__bf16)f1.y;
      BsT[bc + 6][br] = (__bf16)f1.z;
      BsT[bc + 7][br] = (__bf16)f1.w;
    }
    if (k0 + 32 < K) {    // global_prefetch_b8 of next tiles
      __builtin_prefetch(&A[(size_t)(m0 + ar) * lda + (k0 + 32 + ac)], 0, 1);
      __builtin_prefetch(&W[(size_t)(k0 + 32 + br) * N + (n0 + bc)], 0, 1);
    }
    __syncthreads();

    // ---- fragments per wave32 VGPR layouts (ISA 7.12.2), all ds_load_b128
    // A 16x32: lanes 0-15 K0..7 & K16..23 ; lanes 16-31 K8..15 & K24..31
    const int am = wm * 16 + lr;
    const int ak = hi * 8;
    V16 a, b0, b1;
    a.h[0] = *(const v8bf*)&As[am][ak];
    a.h[1] = *(const v8bf*)&As[am][ak + 16];
    // B 32x16: lanes 0-15 K0..7 & K16..23 ; lanes 16-31 K8..15 & K24..31
    const int bn = wn * 32 + lr;
    const int bk = hi * 8;
    b0.h[0] = *(const v8bf*)&BsT[bn][bk];
    b0.h[1] = *(const v8bf*)&BsT[bn][bk + 16];
    b1.h[0] = *(const v8bf*)&BsT[bn + 16][bk];
    b1.h[1] = *(const v8bf*)&BsT[bn + 16][bk + 16];

    acc0 = __builtin_amdgcn_wmma_f32_16x16x32_bf16(false, a.v, false, b0.v,
                                                   (short)0, acc0, false, false);
    acc1 = __builtin_amdgcn_wmma_f32_16x16x32_bf16(false, a.v, false, b1.v,
                                                   (short)0, acc1, false, false);
    __syncthreads();
  }

  // Epilogue: C/D layout — VGPR r holds row r + 8*hi, col = lane%16.
  const int col0 = n0 + wn * 32 + lr;
  const int row0 = m0 + wm * 16 + hi * 8;
  #pragma unroll
  for (int r = 0; r < 8; ++r) {
    const int gr = row0 + r;
    {
      float v = acc0[r];
      if (bias) v += bias[col0];
      v = act_apply(v, act);
      if (res) v += res[(size_t)gr * N + col0];
      C[(size_t)gr * N + col0] = v;
    }
    {
      const int c1 = col0 + 16;
      float v = acc1[r];
      if (bias) v += bias[c1];
      v = act_apply(v, act);
      if (res) v += res[(size_t)gr * N + c1];
      C[(size_t)gr * N + c1] = v;
    }
  }
}

// ---------------- LayerNorm over width Wd (one block per row) ----------------
__global__ __launch_bounds__(256) void layernorm_k(
    const float* __restrict__ x, const float* __restrict__ w,
    const float* __restrict__ b, float* __restrict__ y, int Wd)
{
  const int row = blockIdx.x;
  const float* xr = x + (size_t)row * Wd;
  __shared__ float s1[256], s2[256];
  float a = 0.f, q = 0.f;
  for (int j = threadIdx.x; j < Wd; j += 256) { float v = xr[j]; a += v; q += v * v; }
  s1[threadIdx.x] = a; s2[threadIdx.x] = q;
  __syncthreads();
  for (int off = 128; off > 0; off >>= 1) {
    if (threadIdx.x < off) { s1[threadIdx.x] += s1[threadIdx.x + off];
                             s2[threadIdx.x] += s2[threadIdx.x + off]; }
    __syncthreads();
  }
  const float inv = 1.0f / (float)Wd;
  const float mu  = s1[0] * inv;
  const float var = s2[0] * inv - mu * mu;
  const float rs  = rsqrtf(var + 1e-5f);
  for (int j = threadIdx.x; j < Wd; j += 256)
    y[(size_t)row * Wd + j] = (xr[j] - mu) * rs * w[j] + b[j];
}

// ---------------- Depthwise causal conv (DC=4) + bias + SiLU ----------------
// xz: [B,N,2*DI] (reads first half); out xh: [B,N,DI]
__global__ __launch_bounds__(256) void conv_silu_k(
    const float* __restrict__ xz, const float* __restrict__ cw,
    const float* __restrict__ cb, float* __restrict__ xh)
{
  size_t i = (size_t)blockIdx.x * 256 + threadIdx.x;
  if (i >= (size_t)BB * SEQ * DI_) return;
  int d = (int)(i % DI_);
  size_t r = i / DI_;
  int n = (int)(r % SEQ);
  int b = (int)(r / SEQ);
  float s = 0.f;
  #pragma unroll
  for (int j = 0; j < DC_; ++j) {
    int nn = n + j - (DC_ - 1);
    if (nn >= 0)
      s += cw[d * DC_ + j] * xz[((size_t)b * SEQ + nn) * (2 * DI_) + d];
  }
  s += cb[d];
  xh[i] = s * (1.0f / (1.0f + __expf(-s)));   // SiLU
}

// ---------------- Selective scan: 1024 sequential steps ----------------
// One thread per channel d; state h[16] in registers; B/C broadcast via LDS.
// Writes fused gated output: (y + dd*x) * silu(z)
__global__ __launch_bounds__(128) void ssm_scan_k(
    const float* __restrict__ dt,   // [B,N,DI] softplus'd
    const float* __restrict__ xh,   // [B,N,DI]
    const float* __restrict__ dbl,  // [B,N,64]: [.,32:48]=B, [.,48:64]=C
    const float* __restrict__ xz,   // [B,N,2*DI]: z = [.,DI:]
    const float* __restrict__ alog, // [DI,DS]
    const float* __restrict__ dd,   // [DI]
    float* __restrict__ gated)      // [B,N,DI]
{
  const int d = blockIdx.x * 128 + threadIdx.x;
  const int b = blockIdx.y;
  float Ad[DS_];
  #pragma unroll
  for (int s = 0; s < DS_; ++s) Ad[s] = -__expf(alog[(size_t)d * DS_ + s]);
  const float ddv = dd[d];
  float h[DS_];
  #pragma unroll
  for (int s = 0; s < DS_; ++s) h[s] = 0.f;

  __shared__ float Bs[DS_], Cs[DS_];
  for (int n = 0; n < SEQ; ++n) {
    const size_t rb = (size_t)b * SEQ + n;
    if (threadIdx.x < DS_)            Bs[threadIdx.x]       = dbl[rb * 64 + DTR_ + threadIdx.x];
    else if (threadIdx.x < 2 * DS_)   Cs[threadIdx.x - DS_] = dbl[rb * 64 + DTR_ + DS_ + (threadIdx.x - DS_)];
    __syncthreads();
    const float dtv = dt[rb * DI_ + d];
    const float xv  = xh[rb * DI_ + d];
    float y = 0.f;
    #pragma unroll
    for (int s = 0; s < DS_; ++s) {
      h[s] = __expf(dtv * Ad[s]) * h[s] + (dtv * xv) * Bs[s];
      y += h[s] * Cs[s];
    }
    const float zv = xz[rb * (2 * DI_) + DI_ + d];
    gated[rb * DI_ + d] = (y + ddv * xv) * (zv / (1.0f + __expf(-zv)));
    __syncthreads();
  }
}

// ---------------- flip along sequence: y[b,N-1-n,:] = x[b,n,:] ----------------
__global__ __launch_bounds__(256) void flip_k(const float* __restrict__ x,
                                              float* __restrict__ y)
{
  size_t i = (size_t)blockIdx.x * 256 + threadIdx.x;
  if (i >= (size_t)BB * SEQ * DM_) return;
  int d = (int)(i % DM_);
  size_t r = i / DM_;
  int n = (int)(r % SEQ);
  int b = (int)(r / SEQ);
  y[((size_t)b * SEQ + (SEQ - 1 - n)) * DM_ + d] = x[i];
}

// ---------------- attention score tail: sc[row] = s1[row,:256] . aw2 + ab2 ----
__global__ __launch_bounds__(256) void score_k(const float* __restrict__ s1,
                                               const float* __restrict__ aw2,
                                               const float* __restrict__ ab2,
                                               float* __restrict__ sc)
{
  int row = blockIdx.x * 256 + threadIdx.x;
  if (row >= ROWS) return;
  const float* r = s1 + (size_t)row * (DM_ / 2);
  float acc = ab2[0];
  for (int j = 0; j < DM_ / 2; ++j) acc += r[j] * aw2[j];
  sc[row] = acc;
}

// ---------------- softmax over N per batch ----------------
__global__ __launch_bounds__(256) void softmax_k(const float* __restrict__ sc,
                                                 float* __restrict__ at)
{
  const int b = blockIdx.x;
  const float* s = sc + (size_t)b * SEQ;
  __shared__ float red[256];
  float m = -1e30f;
  for (int j = threadIdx.x; j < SEQ; j += 256) m = fmaxf(m, s[j]);
  red[threadIdx.x] = m; __syncthreads();
  for (int off = 128; off > 0; off >>= 1) {
    if (threadIdx.x < off) red[threadIdx.x] = fmaxf(red[threadIdx.x], red[threadIdx.x + off]);
    __syncthreads();
  }
  m = red[0]; __syncthreads();
  float sum = 0.f;
  for (int j = threadIdx.x; j < SEQ; j += 256) sum += __expf(s[j] - m);
  red[threadIdx.x] = sum; __syncthreads();
  for (int off = 128; off > 0; off >>= 1) {
    if (threadIdx.x < off) red[threadIdx.x] += red[threadIdx.x + off];
    __syncthreads();
  }
  const float inv = 1.0f / red[0];
  for (int j = threadIdx.x; j < SEQ; j += 256)
    at[(size_t)b * SEQ + j] = __expf(s[j] - m) * inv;
}

// ---------------- pooled z[b,d] = sum_n at[b,n] * h[b,n,d] ----------------
__global__ __launch_bounds__(256) void pool_k(const float* __restrict__ at,
                                              const float* __restrict__ h,
                                              float* __restrict__ z)
{
  const int d = blockIdx.x * 256 + threadIdx.x;   // grid.x = 2 -> 512
  const int b = blockIdx.y;
  float acc = 0.f;
  for (int n = 0; n < SEQ; ++n)
    acc += at[(size_t)b * SEQ + n] * h[((size_t)b * SEQ + n) * DM_ + d];
  z[(size_t)b * DM_ + d] = acc;
}

// ---------------- final LN over concat(zf,zb) (width 1024) ----------------
__global__ __launch_bounds__(256) void final_ln_k(const float* __restrict__ zf,
                                                  const float* __restrict__ zb,
                                                  const float* __restrict__ nw,
                                                  const float* __restrict__ nb,
                                                  float* __restrict__ out)
{
  const int b = blockIdx.x;
  __shared__ float s1[256], s2[256];
  float a = 0.f, q = 0.f;
  for (int j = threadIdx.x; j < 2 * DM_; j += 256) {
    float v = (j < DM_) ? zf[(size_t)b * DM_ + j] : zb[(size_t)b * DM_ + j - DM_];
    a += v; q += v * v;
  }
  s1[threadIdx.x] = a; s2[threadIdx.x] = q; __syncthreads();
  for (int off = 128; off > 0; off >>= 1) {
    if (threadIdx.x < off) { s1[threadIdx.x] += s1[threadIdx.x + off];
                             s2[threadIdx.x] += s2[threadIdx.x + off]; }
    __syncthreads();
  }
  const float inv = 1.0f / (float)(2 * DM_);
  const float mu  = s1[0] * inv;
  const float var = s2[0] * inv - mu * mu;
  const float rs  = rsqrtf(var + 1e-5f);
  for (int j = threadIdx.x; j < 2 * DM_; j += 256) {
    float v = (j < DM_) ? zf[(size_t)b * DM_ + j] : zb[(size_t)b * DM_ + j - DM_];
    out[(size_t)b * (2 * DM_) + j] = (v - mu) * rs * nw[j] + nb[j];
  }
}

// ---------------- average of attention maps ----------------
__global__ __launch_bounds__(256) void avg_k(const float* __restrict__ a,
                                             const float* __restrict__ b,
                                             float* __restrict__ o)
{
  int i = blockIdx.x * 256 + threadIdx.x;
  if (i < BB * SEQ) o[i] = 0.5f * (a[i] + b[i]);
}

// ---------------- host helpers ----------------
static inline void gemm(hipStream_t st, const float* A, int lda, const float* W,
                        const float* bias, const float* res, float* C,
                        int M, int N, int K, int act)
{
  dim3 g(N / 64, M / 64), blk(256);
  wmma_gemm_k<<<g, blk, 0, st>>>(A, lda, W, bias, res, C, M, N, K, act);
}

static void stack_dir(hipStream_t st, const float* const* P,
                      float* Wh, float* Wu, float* Wbig, float* Wxc,
                      float* Wdbl, float* Wdt, float* Wg)
{
  for (int l = 0; l < 2; ++l) {
    const float* n1w  = P[0]  + (size_t)l * DM_;
    const float* n1b  = P[1]  + (size_t)l * DM_;
    const float* inw  = P[2]  + (size_t)l * DM_ * 2 * DI_;
    const float* cw   = P[3]  + (size_t)l * DI_ * DC_;
    const float* cb   = P[4]  + (size_t)l * DI_;
    const float* xpw  = P[5]  + (size_t)l * DI_ * (DTR_ + 2 * DS_);
    const float* dtw  = P[6]  + (size_t)l * DTR_ * DI_;
    const float* dtb  = P[7]  + (size_t)l * DI_;
    const float* alog = P[8]  + (size_t)l * DI_ * DS_;
    const float* dd   = P[9]  + (size_t)l * DI_;
    const float* ow   = P[10] + (size_t)l * DI_ * DM_;
    const float* n2w  = P[11] + (size_t)l * DM_;
    const float* n2b  = P[12] + (size_t)l * DM_;
    const float* w1   = P[13] + (size_t)l * DM_ * 4 * DM_;
    const float* b1   = P[14] + (size_t)l * 4 * DM_;
    const float* w2   = P[15] + (size_t)l * 4 * DM_ * DM_;
    const float* b2   = P[16] + (size_t)l * DM_;

    // u = LN(h)
    layernorm_k<<<ROWS, 256, 0, st>>>(Wh, n1w, n1b, Wu, DM_);
    // xz = u @ in_w              [4096,512]x[512,2048]
    gemm(st, Wu, DM_, inw, nullptr, nullptr, Wbig, ROWS, 2 * DI_, DM_, 0);
    // xh = silu(conv(xh) + cb)
    conv_silu_k<<<(BB * SEQ * DI_) / 256, 256, 0, st>>>(Wbig, cw, cb, Wxc);
    // dbl = xh @ xpw             [4096,1024]x[1024,64]
    gemm(st, Wxc, DI_, xpw, nullptr, nullptr, Wdbl, ROWS, DTR_ + 2 * DS_, DI_, 0);
    // dt = softplus(dbl[:,:32] @ dtw + dtb)   (strided A via lda=64)
    gemm(st, Wdbl, DTR_ + 2 * DS_, dtw, dtb, nullptr, Wdt, ROWS, DI_, DTR_, 2);
    // scan + gating
    ssm_scan_k<<<dim3(DI_ / 128, BB), 128, 0, st>>>(Wdt, Wxc, Wdbl, Wbig, alog, dd, Wg);
    // h = h + gated @ ow         [4096,1024]x[1024,512], residual in-place
    gemm(st, Wg, DI_, ow, nullptr, Wh, Wh, ROWS, DM_, DI_, 0);
    // u = LN(h)
    layernorm_k<<<ROWS, 256, 0, st>>>(Wh, n2w, n2b, Wu, DM_);
    // big = gelu(u @ w1 + b1)
    gemm(st, Wu, DM_, w1, b1, nullptr, Wbig, ROWS, 4 * DM_, DM_, 1);
    // h = h + big @ w2 + b2
    gemm(st, Wbig, 4 * DM_, w2, b2, Wh, Wh, ROWS, DM_, 4 * DM_, 0);
  }
}

extern "C" void kernel_launch(void* const* d_in, const int* in_sizes, int n_in,
                              void* d_out, int out_size, void* d_ws, size_t ws_size,
                              hipStream_t stream)
{
  (void)in_sizes; (void)n_in; (void)out_size; (void)ws_size;

  const float* x = (const float*)d_in[0];
  const float* fP[17]; const float* bP[17];
  for (int i = 0; i < 17; ++i) { fP[i] = (const float*)d_in[1 + i];
                                 bP[i] = (const float*)d_in[18 + i]; }
  const float* aw1 = (const float*)d_in[35];
  const float* ab1 = (const float*)d_in[36];
  const float* aw2 = (const float*)d_in[37];
  const float* ab2 = (const float*)d_in[38];
  const float* nw  = (const float*)d_in[39];
  const float* nb  = (const float*)d_in[40];
  float* out = (float*)d_out;

  // workspace carve-out (~117 MB)
  char* ws = (char*)d_ws;
  float* Wh   = (float*)(ws);                       //  8 MB [4096,512]
  float* Wu   = (float*)(ws + ((size_t)8  << 20));  //  8 MB
  float* Wbig = (float*)(ws + ((size_t)16 << 20));  // 32 MB [4096,2048]
  float* Wxc  = (float*)(ws + ((size_t)48 << 20));  // 16 MB [4096,1024]
  float* Wdbl = (float*)(ws + ((size_t)64 << 20));  //  1 MB [4096,64]
  float* Wdt  = (float*)(ws + ((size_t)65 << 20));  // 16 MB
  float* Wg   = (float*)(ws + ((size_t)81 << 20));  // 16 MB
  float* Whf  = (float*)(ws + ((size_t)97 << 20));  //  8 MB
  float* Whb  = (float*)(ws + ((size_t)105 << 20)); //  8 MB
  float* Ws1  = (float*)(ws + ((size_t)113 << 20)); //  4 MB [4096,256]
  char*  sm   = ws + ((size_t)117 << 20);
  float* Wsc  = (float*)(sm);                       // 4096 scores
  float* Waf  = (float*)(sm + (32u << 10));
  float* Wab  = (float*)(sm + (64u << 10));
  float* Wzf  = (float*)(sm + (96u << 10));
  float* Wzb  = (float*)(sm + (128u << 10));

  const size_t hbytes = (size_t)ROWS * DM_ * sizeof(float);

  // ---- forward direction ----
  hipMemcpyAsync(Wh, x, hbytes, hipMemcpyDeviceToDevice, stream);
  stack_dir(stream, fP, Wh, Wu, Wbig, Wxc, Wdbl, Wdt, Wg);
  hipMemcpyAsync(Whf, Wh, hbytes, hipMemcpyDeviceToDevice, stream);

  // ---- backward direction ----
  flip_k<<<(BB * SEQ * DM_) / 256, 256, 0, stream>>>(x, Wh);
  stack_dir(stream, bP, Wh, Wu, Wbig, Wxc, Wdbl, Wdt, Wg);
  flip_k<<<(BB * SEQ * DM_) / 256, 256, 0, stream>>>(Wh, Whb);

  // ---- attention pooling: forward ----
  gemm(stream, Whf, DM_, aw1, ab1, nullptr, Ws1, ROWS, DM_ / 2, DM_, 3);  // tanh
  score_k<<<ROWS / 256, 256, 0, stream>>>(Ws1, aw2, ab2, Wsc);
  softmax_k<<<BB, 256, 0, stream>>>(Wsc, Waf);
  pool_k<<<dim3(DM_ / 256, BB), 256, 0, stream>>>(Waf, Whf, Wzf);

  // ---- attention pooling: backward ----
  gemm(stream, Whb, DM_, aw1, ab1, nullptr, Ws1, ROWS, DM_ / 2, DM_, 3);
  score_k<<<ROWS / 256, 256, 0, stream>>>(Ws1, aw2, ab2, Wsc);
  softmax_k<<<BB, 256, 0, stream>>>(Wsc, Wab);
  pool_k<<<dim3(DM_ / 256, BB), 256, 0, stream>>>(Wab, Whb, Wzb);

  // ---- outputs ----
  final_ln_k<<<BB, 256, 0, stream>>>(Wzf, Wzb, nw, nb, out);               // [4,1024]
  avg_k<<<(BB * SEQ) / 256, 256, 0, stream>>>(Waf, Wab, out + BB * 2 * DM_); // [4,1024]
}
